// DualAttention_71871982731651
// MI455X (gfx1250) — matrix operand order repaired and checked by reading, not compiled, not run on hardware
//
#include <hip/hip_runtime.h>
#include <hip/hip_bf16.h>

typedef __attribute__((ext_vector_type(16))) _Float16 v16h;
typedef __attribute__((ext_vector_type(8)))  _Float16 v8h;
typedef __attribute__((ext_vector_type(8)))  float    v8f;

#define B_   2
#define N_   2048
#define D_   1024
#define H_   16
#define HD_  64
#define ENC_ 8
#define BN_  (B_ * N_)
#define EPS_ 1e-5f

// ---------------------------------------------------------------------------
// WMMA fragment loaders (wave32, v_wmma_f32_16x16x32_f16)
// A (16x32, MxK): lane L holds row M=L%16. halves e=0..7 -> K=k0+sel*8+e,
//                 e=8..15 -> K=k0+16+sel*8+(e-8), sel=L/16. (ISA 7.12.2)
// B (32x16, KxN): lane L holds col N=L%16. same K mapping as A.
// Both become two 16-byte vector loads per lane from a [row][k] layout.
// ---------------------------------------------------------------------------
__device__ __forceinline__ v16h load_frag_a(const _Float16* __restrict__ base,
                                            int ld, int m0, int k0) {
  int lane = threadIdx.x & 31;
  int row  = m0 + (lane & 15);
  int sel  = (lane >> 4) & 1;
  const _Float16* p = base + (size_t)row * ld + k0 + sel * 8;
  v8h lo = *(const v8h*)(p);
  v8h hi = *(const v8h*)(p + 16);
  v16h r;
#pragma unroll
  for (int i = 0; i < 8; ++i) { r[i] = lo[i]; r[i + 8] = hi[i]; }
  return r;
}

__device__ __forceinline__ v16h load_frag_b(const _Float16* __restrict__ base,
                                            int ld, int n0, int k0) {
  int lane = threadIdx.x & 31;
  int row  = n0 + (lane & 15);
  int sel  = (lane >> 4) & 1;
  const _Float16* p = base + (size_t)row * ld + k0 + sel * 8;
  v8h lo = *(const v8h*)(p);
  v8h hi = *(const v8h*)(p + 16);
  v16h r;
#pragma unroll
  for (int i = 0; i < 8; ++i) { r[i] = lo[i]; r[i + 8] = hi[i]; }
  return r;
}

__device__ __forceinline__ v8f wmma_f16(v16h a, v16h b, v8f c) {
  return __builtin_amdgcn_wmma_f32_16x16x32_f16(false, a, false, b,
                                                (short)0, c, false, false);
}

// ---------------------------------------------------------------------------
// Elementwise converts
// ---------------------------------------------------------------------------
__global__ void k_f32_to_f16(const float* __restrict__ in,
                             _Float16* __restrict__ out, int n) {
  int i = blockIdx.x * blockDim.x + threadIdx.x;
  if (i < n) out[i] = (_Float16)in[i];
}

// W is [K x Nc] row-major fp32; Wt is [Nc x K] row-major f16.
__global__ void k_transpose_f16(const float* __restrict__ W,
                                _Float16* __restrict__ Wt, int K, int Nc) {
  int i = blockIdx.x * blockDim.x + threadIdx.x;
  if (i < K * Nc) {
    int n = i / K, k = i - n * K;
    Wt[i] = (_Float16)W[(size_t)k * Nc + n];
  }
}

// ---------------------------------------------------------------------------
// f16 WMMA GEMM: C[M x N] = A[M x K] * Bt[N x K]^T
// One wave per 16x64 output strip: 4 fp32 accumulators, A fragment reused
// across the 4 N-tiles, K unrolled by 64 -> 8 wmma per iteration fed by
// 20 b128 loads. fp32 accumulate; store f16 or f32.
// ---------------------------------------------------------------------------
template <bool STORE_F16>
__global__ void k_gemm_wmma(const _Float16* __restrict__ A, int lda,
                            const _Float16* __restrict__ Bt, int ldb,
                            void* __restrict__ C, int ldc,
                            int M, int N, int K) {
  int wave = blockIdx.x * (blockDim.x >> 5) + (threadIdx.x >> 5);
  int ntn  = N >> 6;                       // 64-wide N strips
  int tm   = wave / ntn;
  int tn   = wave - tm * ntn;
  if (tm >= (M >> 4)) return;
  int m0 = tm << 4, n0 = tn << 6;
  int lane = threadIdx.x & 31;

  v8f acc[4] = {};
  for (int k = 0; k < K; k += 64) {
    if (k + 128 < K) {  // global_prefetch_b8 for upcoming K slices
      __builtin_prefetch(A + (size_t)(m0 + (lane & 15)) * lda + k + 128, 0, 1);
      __builtin_prefetch(Bt + (size_t)(n0 + (lane & 31)) * ldb + k + 128, 0, 1);
      __builtin_prefetch(Bt + (size_t)(n0 + 32 + (lane & 31)) * ldb + k + 128, 0, 1);
    }
    v16h a0 = load_frag_a(A, lda, m0, k);
    v16h a1 = load_frag_a(A, lda, m0, k + 32);
#pragma unroll
    for (int t = 0; t < 4; ++t) {
      v16h b0 = load_frag_b(Bt, ldb, n0 + 16 * t, k);
      v16h b1 = load_frag_b(Bt, ldb, n0 + 16 * t, k + 32);
      acc[t] = wmma_f16(a0, b0, acc[t]);
      acc[t] = wmma_f16(a1, b1, acc[t]);
    }
  }

  int rbase = m0 + ((lane >> 4) << 3);
#pragma unroll
  for (int t = 0; t < 4; ++t) {
    int col = n0 + 16 * t + (lane & 15);
    if (STORE_F16) {
      _Float16* Ch = (_Float16*)C;
#pragma unroll
      for (int r = 0; r < 8; ++r)
        Ch[(size_t)(rbase + r) * ldc + col] = (_Float16)acc[t][r];
    } else {
      float* Cf = (float*)C;
#pragma unroll
      for (int r = 0; r < 8; ++r)
        Cf[(size_t)(rbase + r) * ldc + col] = acc[t][r];
    }
  }
}

// ---------------------------------------------------------------------------
// RoPE + RMSNorm. One wave per (b, n, h). lane = rotation pair (HD/2 == 32).
// Reads f16 QKV [BN][D], writes Qh/Kh [B][H][N][64] and Vt [B][H][64][N].
// ---------------------------------------------------------------------------
__global__ void k_rope_rmsnorm(const _Float16* __restrict__ Qin,
                               const _Float16* __restrict__ Kin,
                               const _Float16* __restrict__ Vin,
                               const float* __restrict__ rc,   // [N][32]
                               const float* __restrict__ rs,   // [N][32]
                               _Float16* __restrict__ Qh,
                               _Float16* __restrict__ Kh,
                               _Float16* __restrict__ Vt) {
  int wave = blockIdx.x * (blockDim.x >> 5) + (threadIdx.x >> 5);
  int h = wave & (H_ - 1);
  int n = (wave >> 4) & (N_ - 1);
  int b = wave >> 15;  // 4 (H) + 11 (N) bits
  int lane = threadIdx.x & 31;

  size_t rowoff = (size_t)(b * N_ + n) * D_ + h * HD_;
  float c = rc[n * 32 + lane];
  float s = rs[n * 32 + lane];

  float qre = (float)Qin[rowoff + 2 * lane];
  float qim = (float)Qin[rowoff + 2 * lane + 1];
  float kre = (float)Kin[rowoff + 2 * lane];
  float kim = (float)Kin[rowoff + 2 * lane + 1];

  float qr = qre * c - qim * s, qi = qre * s + qim * c;
  float kr = kre * c - kim * s, ki = kre * s + kim * c;

  float sq = qr * qr + qi * qi;
  float sk = kr * kr + ki * ki;
#pragma unroll
  for (int m = 16; m >= 1; m >>= 1) {
    sq += __shfl_xor(sq, m, 32);
    sk += __shfl_xor(sk, m, 32);
  }
  float rq = rsqrtf(sq * (1.0f / HD_) + EPS_);
  float rk = rsqrtf(sk * (1.0f / HD_) + EPS_);

  size_t ho = (((size_t)b * H_ + h) * N_ + n) * HD_;
  Qh[ho + 2 * lane]     = (_Float16)(qr * rq);
  Qh[ho + 2 * lane + 1] = (_Float16)(qi * rq);
  Kh[ho + 2 * lane]     = (_Float16)(kr * rk);
  Kh[ho + 2 * lane + 1] = (_Float16)(ki * rk);

  float v0 = (float)Vin[rowoff + 2 * lane];
  float v1 = (float)Vin[rowoff + 2 * lane + 1];
  size_t vo = ((size_t)b * H_ + h) * HD_ * (size_t)N_;
  Vt[vo + (size_t)(2 * lane) * N_ + n]     = (_Float16)v0;
  Vt[vo + (size_t)(2 * lane + 1) * N_ + n] = (_Float16)v1;
}

// ---------------------------------------------------------------------------
// Flash attention. One wave per (b, h, 16-query tile). 32-key blocks.
// Fragment lifetimes are staggered to avoid spills: K fragments die in the
// QK^T WMMAs; V fragments are loaded just before the softmax VALU block
// (which hides their latency) and die in the PV WMMAs.
// __launch_bounds__(256,1): one WG/CU so the allocator can use the full
// VGPR file instead of spilling. Causal heads exit early (wave-uniform).
// ---------------------------------------------------------------------------
__global__ __launch_bounds__(256, 1)
void k_flash_attn(const _Float16* __restrict__ Qh,
                  const _Float16* __restrict__ Kh,
                  const _Float16* __restrict__ Vt,
                  _Float16* __restrict__ O) {
  __shared__ _Float16 sP[8][16 * 32];  // per-wave P tile, 8 KB total
  int widx = threadIdx.x >> 5;
  int wave = blockIdx.x * (blockDim.x >> 5) + widx;
  int qt = wave & 127;          // 128 query tiles
  int hh = (wave >> 7) & 15;    // head
  int b  = wave >> 11;          // batch
  int lane = threadIdx.x & 31;
  int half = lane >> 4;

  const _Float16* Q  = Qh + ((size_t)b * H_ + hh) * N_ * HD_;
  const _Float16* Kp = Kh + ((size_t)b * H_ + hh) * N_ * HD_;
  const _Float16* Vp = Vt + ((size_t)b * H_ + hh) * HD_ * (size_t)N_;
  _Float16* sp = sP[widx];

  int q0 = qt << 4;
  v16h a0 = load_frag_a(Q, HD_, q0, 0);
  v16h a1 = load_frag_a(Q, HD_, q0, 32);

  v8f o0 = {}, o1 = {}, o2 = {}, o3 = {};
  float mrun[8], lrun[8];
#pragma unroll
  for (int r = 0; r < 8; ++r) { mrun[r] = -1e30f; lrun[r] = 0.0f; }

  const bool causal = (hh >= ENC_);
  const int jend = causal ? (q0 + 16) : N_;
  const float scale = 0.125f;  // HD^-0.5

  for (int j0 = 0; j0 < jend; j0 += 32) {
    // ---- S = Q K^T; K fragments are consumed immediately ----
    v8f st[2];
    {
      v16h kb00 = load_frag_b(Kp, HD_, j0,      0);
      v16h kb01 = load_frag_b(Kp, HD_, j0,      32);
      v16h kb10 = load_frag_b(Kp, HD_, j0 + 16, 0);
      v16h kb11 = load_frag_b(Kp, HD_, j0 + 16, 32);
      if (j0 + 32 < jend)  // prefetch next K block rows
        __builtin_prefetch(Kp + (size_t)(j0 + 32 + lane) * HD_, 0, 1);
      v8f cc = {};
      cc = wmma_f16(a0, kb00, cc);
      cc = wmma_f16(a1, kb01, cc);
      st[0] = cc;
      v8f dd = {};
      dd = wmma_f16(a0, kb10, dd);
      dd = wmma_f16(a1, kb11, dd);
      st[1] = dd;
    }

    // ---- issue V fragment loads now; softmax VALU hides their latency ----
    v16h vb0 = load_frag_b(Vp, N_,  0, j0);
    v16h vb1 = load_frag_b(Vp, N_, 16, j0);
    v16h vb2 = load_frag_b(Vp, N_, 32, j0);
    v16h vb3 = load_frag_b(Vp, N_, 48, j0);
    if (j0 + 32 < jend) {  // prefetch next V slices
      __builtin_prefetch(Vp + (size_t)lane * N_ + j0 + 32, 0, 1);
      __builtin_prefetch(Vp + (size_t)(lane + 32) * N_ + j0 + 32, 0, 1);
    }

    // ---- scale + causal mask ----
    int jc = j0 + (lane & 15);
#pragma unroll
    for (int r = 0; r < 8; ++r) {
      int i = q0 + r + 8 * half;
      float s0 = st[0][r] * scale;
      float s1 = st[1][r] * scale;
      if (causal) {
        if (jc > i)      s0 = -1e30f;
        if (jc + 16 > i) s1 = -1e30f;
      }
      st[0][r] = s0;
      st[1][r] = s1;
    }

    // ---- online softmax (per-row reduce across the 16-lane half) ----
#pragma unroll
    for (int r = 0; r < 8; ++r) {
      float mx = fmaxf(st[0][r], st[1][r]);
#pragma unroll
      for (int m = 1; m < 16; m <<= 1) mx = fmaxf(mx, __shfl_xor(mx, m, 32));
      float mnew  = fmaxf(mrun[r], mx);
      float alpha = __expf(mrun[r] - mnew);
      float p0 = __expf(st[0][r] - mnew);
      float p1 = __expf(st[1][r] - mnew);
      st[0][r] = p0;
      st[1][r] = p1;
      float ls = p0 + p1;
#pragma unroll
      for (int m = 1; m < 16; m <<= 1) ls += __shfl_xor(ls, m, 32);
      lrun[r] = lrun[r] * alpha + ls;
      mrun[r] = mnew;
      o0[r] *= alpha; o1[r] *= alpha; o2[r] *= alpha; o3[r] *= alpha;
    }

    // ---- P (C-layout) -> LDS row-major 16x32 -> reload as A fragment ----
#pragma unroll
    for (int r = 0; r < 8; ++r) {
      sp[(r + 8 * half) * 32 + (lane & 15)]      = (_Float16)st[0][r];
      sp[(r + 8 * half) * 32 + 16 + (lane & 15)] = (_Float16)st[1][r];
    }
    asm volatile("s_wait_dscnt 0" ::: "memory");  // wave-local DS RAW fence
    v16h pa = load_frag_a(sp, 32, 0, 0);

    // ---- O += P * V  (V fragments resident; die here) ----
    o0 = wmma_f16(pa, vb0, o0);
    o1 = wmma_f16(pa, vb1, o1);
    o2 = wmma_f16(pa, vb2, o2);
    o3 = wmma_f16(pa, vb3, o3);
  }

  // ---- normalize + store O as f16 [BN][H*HD] ----
  int col = hh * HD_ + (lane & 15);
#pragma unroll
  for (int r = 0; r < 8; ++r) {
    float inv = 1.0f / lrun[r];
    size_t ro = ((size_t)b * N_ + q0 + r + 8 * half) * (size_t)D_;
    O[ro + col +  0] = (_Float16)(o0[r] * inv);
    O[ro + col + 16] = (_Float16)(o1[r] * inv);
    O[ro + col + 32] = (_Float16)(o2[r] * inv);
    O[ro + col + 48] = (_Float16)(o3[r] * inv);
  }
}

// ---------------------------------------------------------------------------
extern "C" void kernel_launch(void* const* d_in, const int* in_sizes, int n_in,
                              void* d_out, int out_size, void* d_ws, size_t ws_size,
                              hipStream_t stream) {
  const float* x   = (const float*)d_in[0];
  const float* rc  = (const float*)d_in[1];
  const float* rs  = (const float*)d_in[2];
  const float* Wq  = (const float*)d_in[3];
  const float* Wk  = (const float*)d_in[4];
  const float* Wv  = (const float*)d_in[5];
  const float* Woe = (const float*)d_in[6];
  const float* Wod = (const float*)d_in[7];
  float* out = (float*)d_out;

  char* ws = (char*)d_ws;
  size_t off = 0;
  auto carve = [&](size_t bytes) {
    void* p = ws + off;
    off = (off + bytes + 255) & ~(size_t)255;
    return p;
  };

  _Float16* Xh   = (_Float16*)carve((size_t)BN_ * D_ * 2);      // 8 MB
  _Float16* Wqt  = (_Float16*)carve((size_t)D_ * D_ * 2);       // 2 MB
  _Float16* Wkt  = (_Float16*)carve((size_t)D_ * D_ * 2);
  _Float16* Wvt  = (_Float16*)carve((size_t)D_ * D_ * 2);
  _Float16* Woet = (_Float16*)carve((size_t)D_ * D_ * 2);
  _Float16* Wodt = (_Float16*)carve((size_t)D_ * 512 * 2);      // 1 MB
  _Float16* Q16  = (_Float16*)carve((size_t)BN_ * D_ * 2);      // 8 MB
  _Float16* K16  = (_Float16*)carve((size_t)BN_ * D_ * 2);
  _Float16* V16  = (_Float16*)carve((size_t)BN_ * D_ * 2);
  _Float16* Qh   = (_Float16*)carve((size_t)B_ * H_ * N_ * HD_ * 2);  // 8 MB
  _Float16* Kh   = (_Float16*)carve((size_t)B_ * H_ * N_ * HD_ * 2);
  _Float16* Vt   = (_Float16*)carve((size_t)B_ * H_ * N_ * HD_ * 2);
  _Float16* Oh   = (_Float16*)carve((size_t)BN_ * D_ * 2);      // 8 MB

  const int TPB = 256;

  // 1) x -> f16
  {
    int n = BN_ * D_;
    k_f32_to_f16<<<(n + TPB - 1) / TPB, TPB, 0, stream>>>(x, Xh, n);
  }
  // 2) transpose weights -> f16 [out][k]
  {
    int n = D_ * D_;
    int g = (n + TPB - 1) / TPB;
    k_transpose_f16<<<g, TPB, 0, stream>>>(Wq,  Wqt,  D_, D_);
    k_transpose_f16<<<g, TPB, 0, stream>>>(Wk,  Wkt,  D_, D_);
    k_transpose_f16<<<g, TPB, 0, stream>>>(Wv,  Wvt,  D_, D_);
    k_transpose_f16<<<g, TPB, 0, stream>>>(Woe, Woet, D_, D_);
    int nd = 512 * D_;
    k_transpose_f16<<<(nd + TPB - 1) / TPB, TPB, 0, stream>>>(Wod, Wodt, 512, D_);
  }
  // 3) QKV projections (f16 out)
  {
    int waves = (BN_ / 16) * (D_ / 64);     // 4096
    int g = waves / (TPB / 32);             // 512 blocks
    k_gemm_wmma<true><<<g, TPB, 0, stream>>>(Xh, D_, Wqt, D_, Q16, D_, BN_, D_, D_);
    k_gemm_wmma<true><<<g, TPB, 0, stream>>>(Xh, D_, Wkt, D_, K16, D_, BN_, D_, D_);
    k_gemm_wmma<true><<<g, TPB, 0, stream>>>(Xh, D_, Wvt, D_, V16, D_, BN_, D_, D_);
  }
  // 4) RoPE + RMSNorm + per-head relayout (V transposed)
  {
    int waves = B_ * N_ * H_;               // 65536
    k_rope_rmsnorm<<<waves / (TPB / 32), TPB, 0, stream>>>(Q16, K16, V16, rc, rs,
                                                           Qh, Kh, Vt);
  }
  // 5) flash attention
  {
    int waves = B_ * H_ * (N_ / 16);        // 4096
    k_flash_attn<<<waves / (TPB / 32), TPB, 0, stream>>>(Qh, Kh, Vt, Oh);
  }
  // 6) output projections (f32 out to d_out)
  {
    int waves = (BN_ / 16) * (D_ / 64);     // 4096
    int g = waves / (TPB / 32);
    k_gemm_wmma<false><<<g, TPB, 0, stream>>>(Oh, D_, Woet, D_, out, D_,
                                              BN_, D_, D_);
    k_gemm_wmma<false><<<g, TPB, 0, stream>>>(Oh + ENC_ * HD_, D_, Wodt, 512,
                                              out + (size_t)BN_ * D_, D_,
                                              BN_, D_, 512);
  }
}
